// AttentionAggregator_25924422598756
// MI455X (gfx1250) — matrix operand order repaired and checked by reading, not compiled, not run on hardware
//
#include <hip/hip_runtime.h>
#include <hip/hip_fp16.h>

typedef _Float16 h4  __attribute__((ext_vector_type(4)));
typedef _Float16 h8  __attribute__((ext_vector_type(8)));
typedef _Float16 v16h __attribute__((ext_vector_type(16)));
typedef float    v8f  __attribute__((ext_vector_type(8)));

constexpr int Bn   = 4096;
constexpr int Hh   = 200;
constexpr int EMB  = 128;
constexpr int HID  = 64;
constexpr int HID2 = 32;
constexpr int HP   = 208;          // H padded to 13*16
constexpr int NTHREADS = 416;      // 13 waves of 32

constexpr int AS  = 136;           // histA / w1t row stride (halves): 272B = 68 dwords -> bank-friendly
constexpr int H1S = 72;            // h1 row stride (halves)
constexpr int W2S = 72;            // w2t row stride (halves)
constexpr int H2S = 33;            // h2 row stride (floats), odd -> conflict-free columns

// LDS layout (bytes)
constexpr int OFF_HISTA = 0;                           // 208*136*2 = 56576
constexpr int OFF_W1T   = OFF_HISTA + HP * AS * 2;     // 64*136*2  = 17408
constexpr int OFF_H1    = OFF_W1T + HID * AS * 2;      // 208*72*2  = 29952
constexpr int OFF_W2T   = OFF_H1 + HP * H1S * 2;       // 32*72*2   = 4608
constexpr int OFF_H2    = OFF_W2T + HID2 * W2S * 2;    // 208*33*4  = 27456
constexpr int OFF_BIAS1 = OFF_H2 + HP * H2S * 4;       // 64*4
constexpr int OFF_B2    = OFF_BIAS1 + 64 * 4;          // 32*4
constexpr int OFF_W3    = OFF_B2 + 32 * 4;             // 32*4
constexpr int OFF_SC    = OFF_W3 + 32 * 4;             // 256*4
constexpr int OFF_RED   = OFF_SC + 256 * 4;            // 256*4
constexpr int OFF_WTS   = OFF_RED + 256 * 4;           // 256*4 (200 read)
constexpr int OFF_AGGP  = OFF_WTS + 256 * 4;           // 3*128*4
constexpr int SMEM_BYTES = OFF_AGGP + 3 * 128 * 4;     // ~141 KB

__global__ __launch_bounds__(NTHREADS)
void attn_agg_wmma(const float* __restrict__ hist, const float* __restrict__ cand,
                   const float* __restrict__ W1, const float* __restrict__ b1,
                   const float* __restrict__ W2, const float* __restrict__ b2,
                   const float* __restrict__ W3, const float* __restrict__ b3,
                   const int* __restrict__ mask,
                   float* __restrict__ out_agg, float* __restrict__ out_attw) {
  extern __shared__ __align__(16) char smem[];
  _Float16* histA = (_Float16*)(smem + OFF_HISTA);
  _Float16* w1t   = (_Float16*)(smem + OFF_W1T);
  _Float16* h1s   = (_Float16*)(smem + OFF_H1);
  _Float16* w2t   = (_Float16*)(smem + OFF_W2T);
  float*    h2s   = (float*)(smem + OFF_H2);
  float*    bias1 = (float*)(smem + OFF_BIAS1);
  float*    b2s   = (float*)(smem + OFF_B2);
  float*    w3s   = (float*)(smem + OFF_W3);
  float*    sc    = (float*)(smem + OFF_SC);
  float*    red   = (float*)(smem + OFF_RED);
  float*    wts   = (float*)(smem + OFF_WTS);
  float*    aggp  = (float*)(smem + OFF_AGGP);

  const int b   = blockIdx.x;
  const int tid = threadIdx.x;

  // --- Stage historical_embs[b] as f16 in LDS (pad rows zeroed). 128-bit global loads.
  const float* hb = hist + (size_t)b * Hh * EMB;
  for (int idx = tid; idx < HP * (EMB / 4); idx += NTHREADS) {
    int m = idx >> 5;            // 32 float4 per 128-wide row
    int c4 = idx & 31;
    float4 v = make_float4(0.f, 0.f, 0.f, 0.f);
    if (m < Hh) v = ((const float4*)hb)[idx];
    h4 t;
    t[0] = (_Float16)v.x; t[1] = (_Float16)v.y; t[2] = (_Float16)v.z; t[3] = (_Float16)v.w;
    *(h4*)(histA + m * AS + c4 * 4) = t;
  }
  // --- W1 (history half, rows 0..127) transposed -> f16
  for (int idx = tid; idx < EMB * HID; idx += NTHREADS) {
    int k = idx >> 6, n = idx & 63;
    w1t[n * AS + k] = (_Float16)W1[k * HID + n];
  }
  // --- W2 transposed -> f16
  for (int idx = tid; idx < HID * HID2; idx += NTHREADS) {
    int k = idx >> 5, n = idx & 31;
    w2t[n * W2S + k] = (_Float16)W2[k * HID2 + n];
  }
  // --- Fold candidate into layer-1 bias: bias1 = b1 + cand[b] @ W1[128:,:]
  if (tid < HID) {
    float s = b1[tid];
    const float* cb = cand + (size_t)b * EMB;
    for (int k = 0; k < EMB; ++k) s += cb[k] * W1[(EMB + k) * HID + tid];
    bias1[tid] = s;
  }
  if (tid >= 64 && tid < 96) { b2s[tid - 64] = b2[tid - 64]; w3s[tid - 64] = W3[tid - 64]; }
  __syncthreads();

  const int wv   = tid >> 5;
  const int lane = tid & 31;
  const int g    = lane >> 4;     // K half-group
  const int ml   = lane & 15;
  const int Mrow = wv * 16 + ml;

  // ================= Layer 1: hist[208x128] @ W1t -> h1[208x64]  (f16 WMMA, f32 acc)
  v8f acc[4];
  #pragma unroll
  for (int nt = 0; nt < 4; ++nt)
    #pragma unroll
    for (int i = 0; i < 8; ++i) acc[nt][i] = 0.f;

  #pragma unroll
  for (int kc = 0; kc < 4; ++kc) {
    const _Float16* ap = histA + Mrow * AS + kc * 32 + 8 * g;
    h8 alo = *(const h8*)ap;
    h8 ahi = *(const h8*)(ap + 16);
    v16h a;
    #pragma unroll
    for (int i = 0; i < 8; ++i) { a[i] = alo[i]; a[8 + i] = ahi[i]; }
    #pragma unroll
    for (int nt = 0; nt < 4; ++nt) {
      const _Float16* bp = w1t + (nt * 16 + ml) * AS + kc * 32 + 8 * g;
      h8 blo = *(const h8*)bp;
      h8 bhi = *(const h8*)(bp + 16);
      v16h bb;
      #pragma unroll
      for (int i = 0; i < 8; ++i) { bb[i] = blo[i]; bb[8 + i] = bhi[i]; }
      acc[nt] = __builtin_amdgcn_wmma_f32_16x16x32_f16(
          false, a, false, bb, (short)0, acc[nt], false, false);
    }
  }
  // relu + folded bias, store h1 as f16
  #pragma unroll
  for (int nt = 0; nt < 4; ++nt) {
    float bn = bias1[nt * 16 + ml];
    #pragma unroll
    for (int r = 0; r < 8; ++r) {
      int mr = wv * 16 + r + 8 * g;
      h1s[mr * H1S + nt * 16 + ml] = (_Float16)fmaxf(acc[nt][r] + bn, 0.f);
    }
  }
  __syncthreads();

  // ================= Layer 2: h1[208x64] @ W2t -> h2[208x32]
  v8f c2[2];
  #pragma unroll
  for (int nt = 0; nt < 2; ++nt)
    #pragma unroll
    for (int i = 0; i < 8; ++i) c2[nt][i] = 0.f;

  #pragma unroll
  for (int kc = 0; kc < 2; ++kc) {
    const _Float16* ap = h1s + Mrow * H1S + kc * 32 + 8 * g;
    h8 alo = *(const h8*)ap;
    h8 ahi = *(const h8*)(ap + 16);
    v16h a;
    #pragma unroll
    for (int i = 0; i < 8; ++i) { a[i] = alo[i]; a[8 + i] = ahi[i]; }
    #pragma unroll
    for (int nt = 0; nt < 2; ++nt) {
      const _Float16* bp = w2t + (nt * 16 + ml) * W2S + kc * 32 + 8 * g;
      h8 blo = *(const h8*)bp;
      h8 bhi = *(const h8*)(bp + 16);
      v16h bb;
      #pragma unroll
      for (int i = 0; i < 8; ++i) { bb[i] = blo[i]; bb[8 + i] = bhi[i]; }
      c2[nt] = __builtin_amdgcn_wmma_f32_16x16x32_f16(
          false, a, false, bb, (short)0, c2[nt], false, false);
    }
  }
  #pragma unroll
  for (int nt = 0; nt < 2; ++nt) {
    float bn = b2s[nt * 16 + ml];
    #pragma unroll
    for (int r = 0; r < 8; ++r) {
      int mr = wv * 16 + r + 8 * g;
      h2s[mr * H2S + nt * 16 + ml] = fmaxf(c2[nt][r] + bn, 0.f);
    }
  }
  __syncthreads();

  // ================= Scores: h2 @ W3 + b3, masked
  if (tid < 256) {
    float s = -1e30f;
    if (tid < Hh) {
      s = b3[0];
      const float* hr = h2s + tid * H2S;
      #pragma unroll
      for (int n = 0; n < HID2; ++n) s += hr[n] * w3s[n];
      s += (1.f - (float)mask[(size_t)b * Hh + tid]) * -1e9f;
    }
    sc[tid] = s;
  }
  __syncthreads();

  // ================= Softmax over H (deterministic tree reductions)
  if (tid < 256) red[tid] = sc[tid];
  __syncthreads();
  for (int st = 128; st > 0; st >>= 1) {
    if (tid < st) red[tid] = fmaxf(red[tid], red[tid + st]);
    __syncthreads();
  }
  float mx = red[0];
  __syncthreads();
  if (tid < 256) {
    float e = (tid < Hh) ? __expf(sc[tid] - mx) : 0.f;
    red[tid] = e;
    wts[tid] = e;                  // 200..255 get 0
  }
  __syncthreads();
  for (int st = 128; st > 0; st >>= 1) {
    if (tid < st) red[tid] += red[tid + st];
    __syncthreads();
  }
  float inv = 1.f / red[0];
  if (tid < Hh) {
    float wf = wts[tid] * inv;
    wts[tid] = wf;
    out_attw[(size_t)b * Hh + tid] = wf;
  }
  __syncthreads();

  // ================= Aggregation: out[d] = sum_h w[h] * hist[h][d] (3-way split, ordered combine)
  int c = tid >> 7, d = tid & 127;
  if (c < 3) {
    int h0 = c * 67;
    int h1e = (h0 + 67 < Hh) ? h0 + 67 : Hh;
    float s = 0.f;
    for (int h = h0; h < h1e; ++h) s += wts[h] * (float)histA[h * AS + d];
    aggp[c * 128 + d] = s;
  }
  __syncthreads();
  if (tid < EMB)
    out_agg[(size_t)b * EMB + tid] = aggp[tid] + aggp[128 + tid] + aggp[256 + tid];
}

extern "C" void kernel_launch(void* const* d_in, const int* in_sizes, int n_in,
                              void* d_out, int out_size, void* d_ws, size_t ws_size,
                              hipStream_t stream) {
  const float* hist = (const float*)d_in[0];
  const float* cand = (const float*)d_in[1];
  const float* W1   = (const float*)d_in[2];
  const float* b1   = (const float*)d_in[3];
  const float* W2   = (const float*)d_in[4];
  const float* b2   = (const float*)d_in[5];
  const float* W3   = (const float*)d_in[6];
  const float* b3   = (const float*)d_in[7];
  const int*   mask = (const int*)d_in[8];

  float* out_agg  = (float*)d_out;
  float* out_attw = out_agg + (size_t)Bn * EMB;

  (void)hipFuncSetAttribute((const void*)attn_agg_wmma,
                            hipFuncAttributeMaxDynamicSharedMemorySize, SMEM_BYTES);

  attn_agg_wmma<<<dim3(Bn), dim3(NTHREADS), SMEM_BYTES, stream>>>(
      hist, cand, W1, b1, W2, b2, W3, b3, mask, out_agg, out_attw);
}
